// CrossModalPairLossWithQueue_68925635166281
// MI455X (gfx1250) — matrix operand order repaired and checked by reading, not compiled, not run on hardware
//
#include <hip/hip_runtime.h>
#include <hip/hip_bf16.h>
#include <math.h>

// ---------------------------------------------------------------------------
// CrossModalPairLossWithQueue on MI455X (gfx1250, wave32, WMMA + TDM)
//
// sizes: B=1024, D=384, Q=65536, N=B+Q=66560, TOPK=128, HARD_W=2
//
// ws layout (256B-aligned sections, ~375 MB total):
//   [0]                 : float inv_temp
//   [256]               : rgb_all_f16  [N, D]  (rows 0..B-1 = normalized rgb feat)
//   [256 + F]           : ir_all_f16   [N, D]
//   [256 + 2F]          : logits f32   [B, N]  (reused for both directions)
//   [256 + 2F + L]      : loss partials [2*B] floats
// ---------------------------------------------------------------------------

#define B_ 1024
#define D_ 384
#define Q_ 65536
#define N_ (B_ + Q_)      // 66560
#define TOPK_ 128
#define NBINS 2048
#define CAP 2048

// GEMM blocking: block = 256 thr = 8 waves; each block owns a 128-row M tile
// and sweeps TN_SWEEP 16-col N tiles whose B panels are TDM-DMA'd into LDS.
#define TN_SWEEP 16
// LDS B panel: 16 rows x 384 halves; TDM pads 16B per 256B chunk ->
// row pitch = 768 + 3*16 = 816 bytes (204 dwords == 12 mod 64 banks).
#define ROW_PITCH 816
#define PANEL_BYTES (16 * ROW_PITCH)   // 13056

typedef _Float16 v16h __attribute__((ext_vector_type(16)));
typedef _Float16 v8h  __attribute__((ext_vector_type(8)));
typedef _Float16 v4h  __attribute__((ext_vector_type(4)));
typedef float    v8f  __attribute__((ext_vector_type(8)));
typedef float    v4f  __attribute__((ext_vector_type(4)));
typedef unsigned int u32x4 __attribute__((ext_vector_type(4)));
typedef int          i32x8 __attribute__((ext_vector_type(8)));
typedef int          i32x4 __attribute__((ext_vector_type(4)));

// --- temperature curriculum (epoch is a device int) ------------------------
__global__ void compute_temp_kernel(const int* __restrict__ epochPtr,
                                    float* __restrict__ invTemp) {
  if (threadIdx.x == 0 && blockIdx.x == 0) {
    int epoch = *epochPtr;
    float t;
    if (epoch >= 10) {
      t = 0.05f;
    } else {
      float prog = (float)epoch / 10.0f;
      t = 0.05f + 0.5f * (0.2f - 0.05f) * (1.0f + cosf(3.14159265358979323846f * prog));
    }
    invTemp[0] = 1.0f / t;
  }
}

// --- L2-normalize a [B_, D_] fp32 matrix, store f16 into dst rows ----------
__global__ __launch_bounds__(128)
void normalize_rows_kernel(const float* __restrict__ src,
                           _Float16* __restrict__ dst) {
  __shared__ float red[128];
  const int row = blockIdx.x;
  const int tid = threadIdx.x;
  const float* p = src + (size_t)row * D_;
  float vals[3];
  float a = 0.f;
#pragma unroll
  for (int c = 0; c < 3; ++c) {
    float v = p[tid + c * 128];
    vals[c] = v;
    a += v * v;
  }
  red[tid] = a;
  __syncthreads();
  for (int off = 64; off > 0; off >>= 1) {
    if (tid < off) red[tid] += red[tid + off];
    __syncthreads();
  }
  float inv = 1.0f / fmaxf(sqrtf(red[0]), 1e-12f);
#pragma unroll
  for (int c = 0; c < 3; ++c)
    dst[(size_t)row * D_ + tid + c * 128] = (_Float16)(vals[c] * inv);
}

// --- fp32 -> fp16 bulk convert (queues are pre-normalized) -----------------
__global__ __launch_bounds__(256)
void convert_f16_kernel(const float* __restrict__ src,
                        _Float16* __restrict__ dst, int count4) {
  int idx = blockIdx.x * 256 + threadIdx.x;
  if (idx < count4) {
    v4f v = *(const v4f*)(src + (size_t)idx * 4);
    v4h o;
#pragma unroll
    for (int e = 0; e < 4; ++e) o[e] = (_Float16)v[e];
    *(v4h*)(dst + (size_t)idx * 4) = o;
  }
}

// --- TDM: DMA one 16-row x 384-half B panel into LDS -----------------------
// D# per CDNA5 ISA 8.3/8.4: count=1, type=2("image"); data_size=1 (2 bytes);
// tensor = [N_, 384] row-major (dim0=384, dim0_stride=384, dim1=N_);
// tile = 384 x 16; LDS padding: 4 dwords after every 64 dwords
// (pad_interval=5, pad_amount=3) -> 816B row pitch in LDS.
// This toolchain declares the 6-arg builtin (u32x4, i32x8, i32x4, i32x4,
// i32x8, i32 cpol); trailing descriptor groups are zero (2-D tensor, no
// gather/iterate/multicast).
__device__ __forceinline__ void tdm_load_panel(const _Float16* gsrc,
                                               unsigned ldsAddr) {
  unsigned long long ga = (unsigned long long)(size_t)gsrc;
  u32x4 g0;
  g0[0] = 1u;                                    // count=1
  g0[1] = ldsAddr;                               // lds_addr [63:32]
  g0[2] = (unsigned)(ga & 0xFFFFFFFFull);        // global_addr [95:64]
  g0[3] = (unsigned)((ga >> 32) & 0x1FFFFFFull)  // global_addr [120:96]
          | 0x80000000u;                         // type=2 -> bits[127:126]
  i32x8 g1;
  g1[0] = (int)((1u << 16)      // data_size = 2 bytes
                | (1u << 20)    // pad_enable
                | (5u << 22)    // pad_interval: every 64 dwords (256B)
                | (3u << 25));  // pad_amount: 4 dwords (16B)
  g1[1] = (int)(384u << 16);                  // tensor_dim0=384 (bits[79:48])
  g1[2] = (int)(((unsigned)N_ & 0xFFFFu) << 16);  // tensor_dim1 lo (bits[111:80])
  g1[3] = (int)((((unsigned)N_ >> 16) & 0xFFFFu)  // tensor_dim1 hi
                | (384u << 16));                  // tile_dim0=384 (bits[127:112])
  g1[4] = 16;                                 // tile_dim1=16, tile_dim2=0
  g1[5] = 384;                                // tensor_dim0_stride lo32
  g1[6] = 0;                                  // stride hi / dim1_stride lo
  g1[7] = 0;
  i32x4 z4 = {0, 0, 0, 0};
  i32x8 z8 = {0, 0, 0, 0, 0, 0, 0, 0};
  __builtin_amdgcn_tensor_load_to_lds(g0, g1, z4, z4, z8, 0);
}

// --- GEMM: out[B_,N_] = (A[B_,D_] @ Bm[N_,D_]^T) * invTemp  (f16 in, f32 acc)
// grid = (N_/16/TN_SWEEP, B_/128) = (260, 8); block = 256 threads = 8 waves.
// Wave w owns M rows [m0+16w, m0+16w+16); all 8 waves share the LDS B panel.
// Per tile: 12 x v_wmma_f32_16x16x32_f16; B fragments via ds_load_b128 from
// the TDM-staged (double-buffered) panel. Fragment layouts used here (contiguous k):
//   A 16x32 f16 : lane<16 (M=lane)     K[0..15];
//                 lane>=16 (M=lane-16) K[16..31]
//   B 32x16 f16 : lane<16 (N=lane)     K[0..15]; lane>=16 (N=lane-16) K[16..31]
//   C/D 16x16 f32: vgpr r -> M = r + (lane<16?0:8), N = lane&15
__global__ __launch_bounds__(256)
void gemm_f16_kernel(const _Float16* __restrict__ A,
                     const _Float16* __restrict__ Bm,
                     float* __restrict__ out,
                     const float* __restrict__ invTempPtr) {
  __shared__ __align__(32) char panels[2 * PANEL_BYTES];

  const int lane = threadIdx.x & 31;
  const int wave = threadIdx.x >> 5;
  const int hi   = (lane < 16) ? 0 : 1;
  const int lr   = lane & 15;
  const int m0   = blockIdx.y * 128 + wave * 16;
  const int nTile0 = blockIdx.x * TN_SWEEP;
  const float invTemp = *invTempPtr;
  const unsigned ldsBase = (unsigned)(size_t)(void*)&panels[0];

  // A fragments for all 12 k-steps, resident for the whole N sweep.
  v16h afrag[12];
  const _Float16* arow = A + (size_t)(m0 + lr) * D_ + hi * 16;
#pragma unroll
  for (int kt = 0; kt < 12; ++kt) {
    const _Float16* p = arow + kt * 32;
    v8h lo = *(const v8h*)(p);
    v8h hh = *(const v8h*)(p + 8);
    v16h a;
#pragma unroll
    for (int e = 0; e < 8; ++e) { a[e] = lo[e]; a[e + 8] = hh[e]; }
    afrag[kt] = a;
  }

  // Prologue: DMA first panel.
  if (wave == 0)
    tdm_load_panel(Bm + (size_t)nTile0 * 16 * D_, ldsBase);

#pragma unroll 1
  for (int t = 0; t < TN_SWEEP; ++t) {
    if (wave == 0) {
      if (t + 1 < TN_SWEEP) {
        // DMA next panel into the other buffer (free since barrier of t-1),
        // then wait until only that new DMA is outstanding => panel t ready.
        tdm_load_panel(Bm + (size_t)(nTile0 + t + 1) * 16 * D_,
                       ldsBase + (unsigned)(((t + 1) & 1) * PANEL_BYTES));
        __builtin_amdgcn_s_wait_tensorcnt(1);
      } else {
        __builtin_amdgcn_s_wait_tensorcnt(0);
      }
    }
    __syncthreads();  // panel t visible to all 8 waves

    const char* panel = &panels[(t & 1) * PANEL_BYTES];
    v16h bfrag[12];
#pragma unroll
    for (int kt = 0; kt < 12; ++kt) {
      int b = kt * 64 + hi * 32;            // unpadded byte offset in row
      int pb = b + ((b >> 8) << 4);         // +16B per 256B chunk (TDM pad)
      bfrag[kt] = *(const v16h*)(panel + lr * ROW_PITCH + pb);
    }

    v8f acc = {};
#pragma unroll
    for (int kt = 0; kt < 12; ++kt)
      acc = __builtin_amdgcn_wmma_f32_16x16x32_f16(
          /*neg_a=*/false, afrag[kt], /*neg_b=*/false, bfrag[kt],
          /*c_mod=*/(short)0, acc, /*reuse_a=*/false, /*reuse_b=*/false);

    const int n0 = (nTile0 + t) * 16;
#pragma unroll
    for (int r = 0; r < 8; ++r) {
      int m = m0 + r + hi * 8;
      int n = n0 + lr;
      out[(size_t)m * N_ + n] = acc[r] * invTemp;
    }
    __syncthreads();  // all waves done reading panel t before its buffer is reused
  }
}

// --- per-row: exact top-128 threshold (radix-bin select + tie count) then
//     hard-negative-weighted logsumexp and diagonal CE term ----------------
__global__ __launch_bounds__(256)
void row_loss_kernel(const float* __restrict__ logits,
                     float* __restrict__ lossOut) {
  __shared__ unsigned hist[NBINS];
  __shared__ float cand[CAP];
  __shared__ unsigned s_ncand;
  __shared__ int s_bstar;
  __shared__ unsigned s_cabove;
  __shared__ float s_diag;
  __shared__ float s_t128;
  __shared__ int s_ntie;
  __shared__ float red_m[256];
  __shared__ float red_s[256];

  const int row = blockIdx.x;
  const int tid = threadIdx.x;
  const float* x = logits + (size_t)row * N_;

  for (int b = tid; b < NBINS; b += 256) hist[b] = 0;
  if (tid == 0) s_ncand = 0;
  __syncthreads();

  // pass 1: order-preserving-bit histogram of negatives; capture diagonal
  for (int j = tid; j < N_; j += 256) {
    float v = x[j];
    if (j == row) { s_diag = v; continue; }
    unsigned u = __float_as_uint(v);
    unsigned key = (u & 0x80000000u) ? ~u : (u | 0x80000000u);
    atomicAdd(&hist[key >> 21], 1u);
  }
  __syncthreads();

  if (tid == 0) {
    unsigned c = 0;
    for (int b = NBINS - 1; b >= 0; --b) {
      unsigned h = hist[b];
      if (c + h >= (unsigned)TOPK_) { s_bstar = b; s_cabove = c; break; }
      c += h;
    }
  }
  __syncthreads();
  const int bstar = s_bstar;
  const int need = TOPK_ - (int)s_cabove;  // in [1,128]

  // pass 2: collect the boundary bin's values
  for (int j = tid; j < N_; j += 256) {
    if (j == row) continue;
    float v = x[j];
    unsigned u = __float_as_uint(v);
    unsigned key = (u & 0x80000000u) ? ~u : (u | 0x80000000u);
    if ((int)(key >> 21) == bstar) {
      unsigned pos = atomicAdd(&s_ncand, 1u);
      if (pos < CAP) cand[pos] = v;
    }
  }
  __syncthreads();
  const int ncand = (int)min(s_ncand, (unsigned)CAP);

  // exact rank of the need-th largest among candidates (+ tie count)
  for (int c = tid; c < ncand; c += 256) {
    float v = cand[c];
    int g = 0, e = 0;
    for (int j2 = 0; j2 < ncand; ++j2) {
      float w = cand[j2];
      g += (w > v);
      e += (w == v);
    }
    if (g < need && g + e >= need) { s_t128 = v; s_ntie = need - g; }
  }
  __syncthreads();
  const float t128 = s_t128;
  const int ntie = s_ntie;

  // pass 3: streaming logsumexp with weight 2 on strict top (> t128, j!=row)
  float m = -INFINITY, s = 0.f;
  for (int j = tid; j < N_; j += 256) {
    float v = x[j];
    float val = (v > t128 && j != row) ? 2.f * v : v;
    if (val > m) {
      s = s * expf(m - val) + 1.f;
      m = val;
    } else {
      s += expf(val - m);
    }
  }
  red_m[tid] = m;
  red_s[tid] = s;
  __syncthreads();
  for (int off = 128; off > 0; off >>= 1) {
    if (tid < off) {
      float ma = red_m[tid], sa = red_s[tid];
      float mb = red_m[tid + off], sb = red_s[tid + off];
      float M = fmaxf(ma, mb);
      float S = sa * expf(ma - M) + sb * expf(mb - M);
      red_m[tid] = M;
      red_s[tid] = S;
    }
    __syncthreads();
  }

  if (tid == 0) {
    // tie correction: ntie elements equal to t128 move from weight 1 -> 2
    float M = red_m[0], S = red_s[0];
    float m2 = fmaxf(M, 2.f * t128);
    float S2 = S * expf(M - m2) +
               (float)ntie * (expf(2.f * t128 - m2) - expf(t128 - m2));
    lossOut[row] = m2 + logf(S2) - s_diag;  // = -(diag - m - log sum)
  }
}

// --- final scalar: 0.5*(mean_r2i + mean_i2r) = sum(all)/2048 ---------------
__global__ __launch_bounds__(256)
void finalize_kernel(const float* __restrict__ loss, float* __restrict__ out) {
  __shared__ float r[256];
  float a = 0.f;
  for (int j = threadIdx.x; j < 2 * B_; j += 256) a += loss[j];
  r[threadIdx.x] = a;
  __syncthreads();
  for (int off = 128; off > 0; off >>= 1) {
    if (threadIdx.x < off) r[threadIdx.x] += r[threadIdx.x + off];
    __syncthreads();
  }
  if (threadIdx.x == 0) out[0] = r[0] / (2.0f * B_);
}

extern "C" void kernel_launch(void* const* d_in, const int* in_sizes, int n_in,
                              void* d_out, int out_size, void* d_ws, size_t ws_size,
                              hipStream_t stream) {
  const float* rgb_feat  = (const float*)d_in[0];  // [B, D]
  const float* ir_feat   = (const float*)d_in[1];  // [B, D]
  const float* rgb_queue = (const float*)d_in[2];  // [Q, D] pre-normalized
  const float* ir_queue  = (const float*)d_in[3];  // [Q, D] pre-normalized
  const int*   epoch     = (const int*)d_in[4];
  float* out = (float*)d_out;

  const size_t f16Bytes    = (size_t)N_ * D_ * sizeof(_Float16); // 51,118,080 (256-aligned)
  const size_t logitsBytes = (size_t)B_ * N_ * sizeof(float);    // 272,629,760

  char* ws = (char*)d_ws;
  float*     invTemp = (float*)ws;
  _Float16*  rgbAll  = (_Float16*)(ws + 256);
  _Float16*  irAll   = (_Float16*)(ws + 256 + f16Bytes);
  float*     logits  = (float*)(ws + 256 + 2 * f16Bytes);
  float*     lossArr = (float*)(ws + 256 + 2 * f16Bytes + logitsBytes);

  compute_temp_kernel<<<1, 64, 0, stream>>>(epoch, invTemp);

  // normalized fp32 features -> f16, placed as rows [0, B) of the concat mats
  normalize_rows_kernel<<<B_, 128, 0, stream>>>(rgb_feat, rgbAll);
  normalize_rows_kernel<<<B_, 128, 0, stream>>>(ir_feat, irAll);

  // queues -> f16, rows [B, N)
  const int count4 = (Q_ * D_) / 4;                 // 6,291,456
  const int cgrid = (count4 + 255) / 256;
  convert_f16_kernel<<<cgrid, 256, 0, stream>>>(rgb_queue, rgbAll + (size_t)B_ * D_, count4);
  convert_f16_kernel<<<cgrid, 256, 0, stream>>>(ir_queue,  irAll  + (size_t)B_ * D_, count4);

  dim3 ggrid((N_ / 16) / TN_SWEEP, B_ / 128);  // (260, 8)

  // direction 1: rgb_z @ ir_all^T
  gemm_f16_kernel<<<ggrid, 256, 0, stream>>>(rgbAll, irAll, logits, invTemp);
  row_loss_kernel<<<B_, 256, 0, stream>>>(logits, lossArr);

  // direction 2: ir_z @ rgb_all^T (reuse logits buffer)
  gemm_f16_kernel<<<ggrid, 256, 0, stream>>>(irAll, rgbAll, logits, invTemp);
  row_loss_kernel<<<B_, 256, 0, stream>>>(logits, lossArr + B_);

  finalize_kernel<<<1, 256, 0, stream>>>(lossArr, out);
}
